// SASA_Layer_11527692222590
// MI455X (gfx1250) — matrix operand order repaired and checked by reading, not compiled, not run on hardware
//
#include <hip/hip_runtime.h>
#include <hip/hip_bf16.h>
#include <math.h>

typedef __attribute__((ext_vector_type(16))) _Float16 v16h;
typedef __attribute__((ext_vector_type(8)))  _Float16 v8h;
typedef __attribute__((ext_vector_type(8)))  float    v8f;

#define CC    128
#define HH    56
#define HW    3136      // 56*56
#define NPOS  12544     // 4*3136
#define XS_STRIDE 144   // 128 + 16 halfs -> 288B row, 32B aligned fragments

// ---------------------------------------------------------------------------
// Kernel 0: convert q_w/k_w/v_w (f32 128x128 each) to f16 contiguous buffer,
// and pre-reduce rel_h (64,7,1) / rel_w (64,1,7) over channels -> rel[14].
// ---------------------------------------------------------------------------
__global__ __launch_bounds__(256) void sasa_prep(
    const float* __restrict__ qw, const float* __restrict__ kw,
    const float* __restrict__ vw, const float* __restrict__ rel_h,
    const float* __restrict__ rel_w, _Float16* __restrict__ Wh,
    float* __restrict__ rel)
{
    int t = blockIdx.x * 256 + threadIdx.x;
    if (t < 16384)       Wh[t] = (_Float16)qw[t];
    else if (t < 32768)  Wh[t] = (_Float16)kw[t - 16384];
    else if (t < 49152)  Wh[t] = (_Float16)vw[t - 32768];
    if (blockIdx.x == 0 && threadIdx.x < 14) {
        const float* src = (threadIdx.x < 7) ? rel_h : rel_w;
        int i = (threadIdx.x < 7) ? threadIdx.x : threadIdx.x - 7;
        float s = 0.f;
        #pragma unroll 8
        for (int ch = 0; ch < 64; ++ch) s += src[ch * 7 + i];
        rel[threadIdx.x] = s;
    }
}

// ---------------------------------------------------------------------------
// Kernel 1: fused QKV 1x1-conv GEMM with WMMA f32<-f16 16x16x32.
// Block = 256 threads (8 wave32), tile = 64 positions x 128 out channels.
// Loop order pt -> ks -> mat so each LDS B fragment is loaded exactly once
// and feeds 3 independent WMMAs (Q/K/V share the activation tile).
// ---------------------------------------------------------------------------
__global__ __launch_bounds__(256) void sasa_qkv_gemm(
    const float* __restrict__ x,
    const float* __restrict__ qb, const float* __restrict__ kb,
    const float* __restrict__ vb,
    const _Float16* __restrict__ Wh,   // [3][128][128] f16, row-major
    float* __restrict__ Qo, float* __restrict__ Ko, float* __restrict__ Vo)
{
    __shared__ alignas(32) _Float16 Xs[64 * XS_STRIDE]; // [pos][chan], f16

    const int t   = threadIdx.x;
    const int P0  = blockIdx.x * 64;        // 3136 % 64 == 0 -> never crosses batch
    const int b   = P0 / HW;
    const int hw0 = P0 - b * HW;

    // Stage 64 positions x 128 channels, f32 -> f16, transposed to [pos][chan].
    const float* xb = x + (size_t)b * CC * HW + hw0;
    for (int i = t; i < 8192; i += 256) {
        int c = i >> 6;          // channel
        int p = i & 63;          // local position (coalesced over p)
        Xs[p * XS_STRIDE + c] = (_Float16)xb[(size_t)c * HW + p];
    }
    __syncthreads();

    const int lane = t & 31;
    const int wave = t >> 5;
    const int m  = lane & 15;    // A row / D column index
    const int hi = lane >> 4;    // lane-half selector
    const int o0 = wave * 16;    // this wave's 16 output channels

    const float* biases[3] = { qb, kb, vb };
    float*       outs[3]   = { Qo, Ko, Vo };

    // Preload ALL A fragments: [mat][kstep], 96 VGPRs.
    // Lane(m,hi): halfs[0..15]=K(c0+hi*16..+15)
    v16h afr[3][4];
    float bias8[3][8];
    #pragma unroll
    for (int mat = 0; mat < 3; ++mat) {
        const _Float16* wrow = Wh + mat * 16384 + (o0 + m) * CC;
        #pragma unroll
        for (int ks = 0; ks < 4; ++ks) {
            int c0 = ks * 32;
            v8h lo  = *(const v8h*)(wrow + c0 + hi * 16);
            v8h hi8 = *(const v8h*)(wrow + c0 + hi * 16 + 8);
            v16h a;
            #pragma unroll
            for (int e = 0; e < 8; ++e) { a[e] = lo[e]; a[e + 8] = hi8[e]; }
            afr[mat][ks] = a;
        }
        #pragma unroll
        for (int j = 0; j < 8; ++j) bias8[mat][j] = biases[mat][o0 + hi * 8 + j];
    }

    #pragma unroll
    for (int pt = 0; pt < 4; ++pt) {
        v8f acc[3];
        #pragma unroll
        for (int mat = 0; mat < 3; ++mat)
            #pragma unroll
            for (int j = 0; j < 8; ++j) acc[mat][j] = bias8[mat][j]; // D = W*X + b

        const _Float16* xrow = Xs + (pt * 16 + m) * XS_STRIDE;
        #pragma unroll
        for (int ks = 0; ks < 4; ++ks) {
            // B fragment loaded ONCE: lane(n=m,khi=hi): 16 contiguous K halfs (32B)
            v16h bfr = *(const v16h*)(xrow + ks * 32 + hi * 16);
            #pragma unroll
            for (int mat = 0; mat < 3; ++mat)
                acc[mat] = __builtin_amdgcn_wmma_f32_16x16x32_f16(
                    false, afr[mat][ks], false, bfr, (short)0, acc[mat],
                    false, false);
        }
        #pragma unroll
        for (int mat = 0; mat < 3; ++mat) {
            float* outp = outs[mat] + (size_t)b * CC * HW + hw0 + pt * 16 + m;
            #pragma unroll
            for (int j = 0; j < 8; ++j)
                outp[(size_t)(o0 + hi * 8 + j) * HW] = acc[mat][j];
        }
    }
}

// ---------------------------------------------------------------------------
// Kernel 2: attention in "U-space". One wave32 per 16-element group.
//   qk[kk] = sum_d q(U)*K(U,kk) + qsum*(relh[ki]+relw[kj]); softmax(49);
//   out(U) = sum_kk w[kk]*V(U,kk).  Padding taps fall back to k_b/v_b.
// ---------------------------------------------------------------------------
__global__ __launch_bounds__(256) void sasa_attn(
    const float* __restrict__ Qm, const float* __restrict__ Km,
    const float* __restrict__ Vm,
    const float* __restrict__ kb, const float* __restrict__ vb,
    const float* __restrict__ rel,     // [0..6]=relh_sum, [7..13]=relw_sum
    float* __restrict__ out)
{
    const int wid  = blockIdx.x * 8 + (threadIdx.x >> 5);   // group id, [0,100352)
    const int lane = threadIdx.x & 31;

    const int b    = wid / 25088;             // 8*3136 groups per batch
    int r          = wid - b * 25088;
    const int head = r / HW;
    const int g    = r - head * HW;           // group within head, [0,3136)
    const int cc   = g / 196;                 // 3136/16 = 196 groups per channel
    const int rem0 = (g - cc * 196) * 16;     // base of h*56+w for this group
    const int c    = head * 16 + cc;
    const size_t plane = (size_t)(b * CC + c) * HW;

    // lane's "d" position (lanes 16..31 mirror lanes 0..15)
    const int dl = lane & 15;
    const int rl = rem0 + dl;                 // = h*56 + w
    const int hL = rl / HH;
    const int wL = rl - hL * HH;

    const float qv  = Qm[plane + rl];
    const float kbc = kb[c];
    const float vbc = vb[c];

    // qsum over the 16 d's
    float qs = (lane < 16) ? qv : 0.f;
    #pragma unroll
    for (int off = 16; off >= 1; off >>= 1) qs += __shfl_xor(qs, off, 32);

    // each lane owns kk0=lane and kk1=lane+32 (valid when <49)
    const int  kk1v = (lane + 32 < 49);
    const int  ki0 = lane / 7,        kj0 = lane - ki0 * 7;
    const int  ki1 = (lane + 32) / 7, kj1 = (lane + 32) - ki1 * 7;
    float a0 = 0.f, a1 = 0.f;
    #pragma unroll 4
    for (int d = 0; d < 16; ++d) {
        float qd = __shfl(qv, d, 32);
        int   hd = __shfl(hL, d, 32);
        int   wd = __shfl(wL, d, 32);
        int y0 = hd + ki0 - 3, x0 = wd + kj0 - 3;
        float kv0 = (y0 >= 0 && y0 < HH && x0 >= 0 && x0 < HH)
                        ? Km[plane + y0 * HH + x0] : kbc;
        a0 += qd * kv0;
        int y1 = hd + ki1 - 3, x1 = wd + kj1 - 3;
        float kv1 = (kk1v && y1 >= 0 && y1 < HH && x1 >= 0 && x1 < HH)
                        ? Km[plane + y1 * HH + x1] : kbc;
        a1 += qd * kv1;
    }
    a0 += qs * (rel[ki0] + rel[7 + kj0]);
    a1 = kk1v ? (a1 + qs * (rel[ki1] + rel[7 + kj1])) : -INFINITY;

    // softmax over 49 values spread across the wave
    float mx = fmaxf(a0, a1);
    #pragma unroll
    for (int off = 16; off >= 1; off >>= 1) mx = fmaxf(mx, __shfl_xor(mx, off, 32));
    float e0 = __expf(a0 - mx);
    float e1 = kk1v ? __expf(a1 - mx) : 0.f;
    float s  = e0 + e1;
    #pragma unroll
    for (int off = 16; off >= 1; off >>= 1) s += __shfl_xor(s, off, 32);
    float inv = 1.f / s;
    float w0 = e0 * inv, w1 = e1 * inv;

    // out(d) = sum_kk w[kk] * V(d,kk); lanes 0..15 write
    float acc = 0.f;
    #pragma unroll 7
    for (int kk = 0; kk < 49; ++kk) {
        float wv = (kk < 32) ? __shfl(w0, kk, 32) : __shfl(w1, kk - 32, 32);
        int ki = kk / 7, kj = kk - ki * 7;
        int y = hL + ki - 3, xx = wL + kj - 3;
        float vv = (y >= 0 && y < HH && xx >= 0 && xx < HH)
                       ? Vm[plane + y * HH + xx] : vbc;
        acc += wv * vv;
    }
    if (lane < 16) out[plane + rl] = acc;
}

// ---------------------------------------------------------------------------
extern "C" void kernel_launch(void* const* d_in, const int* in_sizes, int n_in,
                              void* d_out, int out_size, void* d_ws, size_t ws_size,
                              hipStream_t stream)
{
    const float* x     = (const float*)d_in[0];
    const float* q_w   = (const float*)d_in[1];
    const float* q_b   = (const float*)d_in[2];
    const float* k_w   = (const float*)d_in[3];
    const float* k_b   = (const float*)d_in[4];
    const float* v_w   = (const float*)d_in[5];
    const float* v_b   = (const float*)d_in[6];
    const float* rel_h = (const float*)d_in[7];
    const float* rel_w = (const float*)d_in[8];

    char* ws = (char*)d_ws;
    _Float16* Wh  = (_Float16*)ws;                 // 3*16384 halfs = 98304 B
    float*    rel = (float*)(ws + 98304);          // 14 floats (+pad to 98368)
    float*    Qm  = (float*)(ws + 98368);          // 4*128*3136 f32 each
    float*    Km  = Qm + 1605632;
    float*    Vm  = Km + 1605632;

    sasa_prep<<<192, 256, 0, stream>>>(q_w, k_w, v_w, rel_h, rel_w, Wh, rel);
    sasa_qkv_gemm<<<NPOS / 64, 256, 0, stream>>>(x, q_b, k_b, v_b, Wh, Qm, Km, Vm);
    sasa_attn<<<NPOS, 256, 0, stream>>>(Qm, Km, Vm, k_b, v_b, rel, (float*)d_out);
}
